// CrossDeformableAttentionBlockwFlow_30253749633354
// MI455X (gfx1250) — matrix operand order repaired and checked by reading, not compiled, not run on hardware
//
#include <hip/hip_runtime.h>
#include <hip/hip_bf16.h>
#include <math.h>

typedef __attribute__((ext_vector_type(16))) __bf16 v16bf;
typedef __attribute__((ext_vector_type(8)))  float  v8f;

#define DEVINL __device__ __forceinline__

constexpr int kH = 128, kW = 128, kHW = kH * kW, kB = 2, kC = 96;
constexpr int AS_STRIDE = 40;   // 80B row stride: 16B aligned, 20-bank stride (conflict free)
constexpr int BS_STRIDE = 40;
constexpr float kSlope = 0.25f;

// ---------------------------------------------------------------- helpers ---
DEVINL float act_apply(float v, int act) {
  if (act == 1) return v >= 0.f ? v : kSlope * v;
  if (act == 2) {
    float c = 0.7978845608028654f * (v + 0.044715f * v * v * v);
    return 0.5f * v * (1.f + tanhf(c));
  }
  return v;
}

// CDNA5 async global->LDS: per-lane 16B contiguous copy, tracked by ASYNCcnt.
DEVINL void async_copy_b128(const __bf16* gsrc, __bf16* lds_dst) {
  unsigned ldsa = (unsigned)(uintptr_t)lds_dst;  // low 32 bits = LDS byte address
  asm volatile("global_load_async_to_lds_b128 %0, %1, off"
               :: "v"(ldsa), "v"(gsrc) : "memory");
}
DEVINL void async_wait0() {
  asm volatile("s_wait_asynccnt 0x0" ::: "memory");
}

// A-matrix (16x32 bf16) fragment: lanes 0-15 hold K 0..15, lanes 16-31 hold K 16..31
DEVINL v16bf lds_load_a(const __bf16* __restrict__ rowp, int lane) {
  const int b0 = (lane < 16) ? 0 : 16;
  v16bf r;
#pragma unroll
  for (int i = 0; i < 8; i++) { r[i] = rowp[b0 + i]; r[8 + i] = rowp[b0 + 8 + i]; }
  return r;
}
// B-matrix (32x16 bf16) fragment: lanes 0-15 hold K 0..15, lanes 16-31 hold K 16..31
DEVINL v16bf lds_load_b(const __bf16* __restrict__ rowp, int lane) {
  const int k0 = (lane < 16) ? 0 : 16;
  v16bf r;
#pragma unroll
  for (int i = 0; i < 16; i++) r[i] = rowp[k0 + i];
  return r;
}

DEVINL float bilin(const float* __restrict__ img, float py, float px) {
  float y0f = floorf(py), x0f = floorf(px);
  float wy = py - y0f, wx = px - x0f;
  int y0 = (int)y0f, x0 = (int)x0f;
  float v = 0.f;
  if ((unsigned)y0 < (unsigned)kH) {
    if ((unsigned)x0 < (unsigned)kW)       v += img[y0 * kW + x0]     * (1.f - wy) * (1.f - wx);
    if ((unsigned)(x0 + 1) < (unsigned)kW) v += img[y0 * kW + x0 + 1] * (1.f - wy) * wx;
  }
  if ((unsigned)(y0 + 1) < (unsigned)kH) {
    if ((unsigned)x0 < (unsigned)kW)       v += img[(y0 + 1) * kW + x0]     * wy * (1.f - wx);
    if ((unsigned)(x0 + 1) < (unsigned)kW) v += img[(y0 + 1) * kW + x0 + 1] * wy * wx;
  }
  return v;
}

// ------------------------------------------------- 3x3 conv, implicit GEMM --
// D[M, B*HW] = W[M, IC*9] x im2col(X)[IC*9, B*HW], bf16 WMMA, f32 accum.
__global__ __launch_bounds__(256) void k_conv3_wmma(
    const __bf16* __restrict__ Wp, const __bf16* __restrict__ X, long xbs,
    const float* __restrict__ bias, const float* __restrict__ res, long resbs,
    float* __restrict__ Of, long obs, __bf16* __restrict__ Oh, long ohbs,
    int M, int IC, int Kp, int act) {
  __shared__ __bf16 As[64 * AS_STRIDE];
  __shared__ __bf16 Bs[128 * BS_STRIDE];
  const int tid = threadIdx.x;
  const int lane = tid & 31, wid = tid >> 5;
  const int wm = wid & 3, wn = wid >> 2;
  const int n0 = blockIdx.x * 128;
  const int m0 = blockIdx.y * 64;
  const int K = IC * 9;

  const v8f vz = {0.f, 0.f, 0.f, 0.f, 0.f, 0.f, 0.f, 0.f};
  v8f acc[4];
#pragma unroll
  for (int i = 0; i < 4; i++) acc[i] = vz;

  const int ar = tid >> 2, ach = tid & 3;   // A: 64 rows x 4 chunks of 8 bf16
  const int bk = tid & 31, bnc = tid >> 5;  // B: 32 k x 8 chunks of 16 cols

  for (int kc = 0; kc < Kp; kc += 32) {
    // --- A tile: weights [64 x 32], async global->LDS 16B per lane
    const __bf16* asrc = Wp + (size_t)(m0 + ar) * Kp + kc + ach * 8;
    async_copy_b128(asrc, &As[ar * AS_STRIDE + ach * 8]);
    if (kc + 32 < Kp) __builtin_prefetch(asrc + 32, 0, 1);  // global_prefetch_b8
    // --- B tile: im2col gather [128 cols x 32 k]
    const int kk = kc + bk;
    if (kk < K) {
      const int ic = kk / 9, kid = kk - ic * 9;
      const int ky = kid / 3 - 1, kx = kid - (kid / 3) * 3 - 1;
      const __bf16* xp = X + (size_t)ic * kHW;
#pragma unroll
      for (int j = 0; j < 16; j++) {
        const int n = n0 + bnc * 16 + j;
        const int b = n >> 14, hw = n & (kHW - 1);
        const int y = hw >> 7, x = hw & 127;
        const int sy = y + ky, sx = x + kx;
        __bf16 v = (__bf16)0.f;
        if ((unsigned)sy < (unsigned)kH && (unsigned)sx < (unsigned)kW)
          v = xp[(size_t)b * xbs + sy * kW + sx];
        Bs[(bnc * 16 + j) * BS_STRIDE + bk] = v;
      }
    } else {
#pragma unroll
      for (int j = 0; j < 16; j++) Bs[(bnc * 16 + j) * BS_STRIDE + bk] = (__bf16)0.f;
    }
    async_wait0();            // drain this wave's async LDS writes
    __syncthreads();
    const v16bf af = lds_load_a(&As[(wm * 16 + (lane & 15)) * AS_STRIDE], lane);
#pragma unroll
    for (int ns = 0; ns < 4; ns++) {
      const v16bf bfr = lds_load_b(&Bs[(wn * 64 + ns * 16 + (lane & 15)) * BS_STRIDE], lane);
      acc[ns] = __builtin_amdgcn_wmma_f32_16x16x32_bf16(false, af, false, bfr,
                                                        (short)0, acc[ns], false, false);
    }
    __syncthreads();
  }
  const int lm = lane & 15, hi = lane >> 4;
#pragma unroll
  for (int ns = 0; ns < 4; ns++) {
    const int n = n0 + wn * 64 + ns * 16 + lm;
    const int b = n >> 14, hw = n & (kHW - 1);
#pragma unroll
    for (int r = 0; r < 8; r++) {
      const int m = m0 + wm * 16 + hi * 8 + r;
      if (m < M) {
        float v = acc[ns][r];
        if (bias) v += bias[m];
        if (res) v += res[(size_t)b * resbs + (size_t)m * kHW + hw];
        v = act_apply(v, act);
        if (Of) Of[(size_t)b * obs + (size_t)m * kHW + hw] = v;
        if (Oh) Oh[(size_t)b * ohbs + (size_t)m * kHW + hw] = (__bf16)v;
      }
    }
  }
}

// ------------------------------------------------------ generic 1x1 GEMM ---
// col -> (batch = col/cpb, r = col%cpb); X addr = b*xbs + k*xks + r
__global__ __launch_bounds__(256) void k_gemm_wmma(
    const __bf16* __restrict__ Wp, const __bf16* __restrict__ X,
    long xbs, long xks, int cpb,
    const float* __restrict__ bias, const float* __restrict__ res, long resbs,
    float* __restrict__ Of, long obs, long oms,
    __bf16* __restrict__ Oh, long ohbs,
    int M, int K, int Kp, int act) {
  __shared__ __bf16 As[64 * AS_STRIDE];
  __shared__ __bf16 Bs[128 * BS_STRIDE];
  const int tid = threadIdx.x;
  const int lane = tid & 31, wid = tid >> 5;
  const int wm = wid & 3, wn = wid >> 2;
  const int n0 = blockIdx.x * 128;
  const int m0 = blockIdx.y * 64;

  const v8f vz = {0.f, 0.f, 0.f, 0.f, 0.f, 0.f, 0.f, 0.f};
  v8f acc[4];
#pragma unroll
  for (int i = 0; i < 4; i++) acc[i] = vz;

  const int ar = tid >> 2, ach = tid & 3;
  const int bk = tid & 31, bnc = tid >> 5;

  for (int kc = 0; kc < Kp; kc += 32) {
    const __bf16* asrc = Wp + (size_t)(m0 + ar) * Kp + kc + ach * 8;
    async_copy_b128(asrc, &As[ar * AS_STRIDE + ach * 8]);
    if (kc + 32 < Kp) __builtin_prefetch(asrc + 32, 0, 1);
    const int kk = kc + bk;
    if (kk < K) {
#pragma unroll
      for (int j = 0; j < 16; j++) {
        const int n = n0 + bnc * 16 + j;
        const int b = n / cpb;
        const int r = n - b * cpb;
        Bs[(bnc * 16 + j) * BS_STRIDE + bk] = X[(size_t)b * xbs + (size_t)kk * xks + r];
      }
    } else {
#pragma unroll
      for (int j = 0; j < 16; j++) Bs[(bnc * 16 + j) * BS_STRIDE + bk] = (__bf16)0.f;
    }
    async_wait0();
    __syncthreads();
    const v16bf af = lds_load_a(&As[(wm * 16 + (lane & 15)) * AS_STRIDE], lane);
#pragma unroll
    for (int ns = 0; ns < 4; ns++) {
      const v16bf bfr = lds_load_b(&Bs[(wn * 64 + ns * 16 + (lane & 15)) * BS_STRIDE], lane);
      acc[ns] = __builtin_amdgcn_wmma_f32_16x16x32_bf16(false, af, false, bfr,
                                                        (short)0, acc[ns], false, false);
    }
    __syncthreads();
  }
  const int lm = lane & 15, hi = lane >> 4;
#pragma unroll
  for (int ns = 0; ns < 4; ns++) {
    const int n = n0 + wn * 64 + ns * 16 + lm;
    const int b = n / cpb;
    const int r = n - b * cpb;
#pragma unroll
    for (int rr = 0; rr < 8; rr++) {
      const int m = m0 + wm * 16 + hi * 8 + rr;
      if (m < M) {
        float v = acc[ns][rr];
        if (bias) v += bias[m];
        if (res) v += res[(size_t)b * resbs + (size_t)m * oms + r];
        v = act_apply(v, act);
        if (Of) Of[(size_t)b * obs + (size_t)m * oms + r] = v;
        if (Oh) Oh[(size_t)b * ohbs + (size_t)m * oms + r] = (__bf16)v;
      }
    }
  }
}

// ------------------------------------------------------ elementwise/gather --
__global__ void k_pack(const float* __restrict__ w, __bf16* __restrict__ o,
                       int M, int K, int Mp, int Kp) {
  const int i = blockIdx.x * 256 + threadIdx.x;
  if (i >= Mp * Kp) return;
  const int m = i / Kp, k = i - m * Kp;
  o[i] = (m < M && k < K) ? (__bf16)w[(size_t)m * K + k] : (__bf16)0.f;
}

__global__ void k_f2bf(const float* __restrict__ s, __bf16* __restrict__ d, int n) {
  const int i = blockIdx.x * 256 + threadIdx.x;
  if (i < n) d[i] = (__bf16)s[i];
}

__global__ void k_copych48(const __bf16* __restrict__ s, __bf16* __restrict__ d) {
  const int i = blockIdx.x * 256 + threadIdx.x;    // kB*48*kHW
  const int b = i / (48 * kHW), r = i - b * (48 * kHW);
  d[(size_t)b * 96 * kHW + r] = s[(size_t)b * 96 * kHW + r];
}

__global__ void k_bwarp(const float* __restrict__ img, const float* __restrict__ flow,
                        float* __restrict__ out) {
  const int i = blockIdx.x * 256 + threadIdx.x;    // kB*kC*kHW
  const int hw = i & (kHW - 1);
  const int t = i >> 14;
  const int c = t % kC, b = t / kC;
  const int y = hw >> 7, x = hw & 127;
  const float gx = (float)x + flow[((size_t)b * 2 + 0) * kHW + hw];
  const float gy = (float)y + flow[((size_t)b * 2 + 1) * kHW + hw];
  out[i] = bilin(img + ((size_t)b * kC + c) * kHW, gy, gx);
}

__global__ void k_cat1(const float* __restrict__ ftt, const float* __restrict__ wp,
                       const float* __restrict__ fl, __bf16* __restrict__ o) {
  const int i = blockIdx.x * 256 + threadIdx.x;    // kB*194*kHW
  const int hw = i & (kHW - 1);
  const int t = i >> 14;
  const int c = t % 194, b = t / 194;
  float v;
  if (c < 96)       v = ftt[((size_t)b * 96 + c) * kHW + hw];
  else if (c < 192) v = wp[((size_t)b * 96 + (c - 96)) * kHW + hw];
  else              v = fl[((size_t)b * 2 + (c - 192)) * kHW + hw];
  o[i] = (__bf16)v;
}

__global__ void k_qcat(const float* __restrict__ q0, const float* __restrict__ ftt,
                       const float* __restrict__ q1, __bf16* __restrict__ o) {
  const int i = blockIdx.x * 256 + threadIdx.x;    // kB*288*kHW
  const int hw = i & (kHW - 1);
  const int t = i >> 14;
  const int c = t % 288, b = t / 288;
  float v;
  if (c < 96)       v = q0[((size_t)b * 96 + c) * kHW + hw];
  else if (c < 192) v = ftt[((size_t)b * 96 + (c - 96)) * kHW + hw];
  else              v = q1[((size_t)b * 96 + (c - 192)) * kHW + hw];
  o[i] = (__bf16)v;
}

// ref[b][gs][d][hw] = 2*tanh(off[b][2*gs+d][hw]) + ft[b][d][hw]
__global__ void k_ref(const float* __restrict__ off, const float* __restrict__ fl,
                      float* __restrict__ ref) {
  const int i = blockIdx.x * 256 + threadIdx.x;    // kB*108*2*kHW
  const int hw = i & (kHW - 1);
  int t = i >> 14;
  const int d = t & 1; t >>= 1;
  const int gs = t % 108, b = t / 108;
  ref[i] = 2.f * tanhf(off[((size_t)b * 324 + 2 * gs + d) * kHW + hw]) +
           fl[((size_t)b * 2 + d) * kHW + hw];
}

// Deformable conv: per (b,hw,g2) thread computes 12 output channels.
__global__ __launch_bounds__(256) void k_deform(
    const float* __restrict__ feat, const float* __restrict__ ref,
    const float* __restrict__ offm, const float* __restrict__ Wd,
    const float* __restrict__ bd, float* __restrict__ q) {
  __shared__ float ws[96 * 12 * 9];
  for (int i = threadIdx.x; i < 96 * 12 * 9; i += 256) ws[i] = Wd[i];
  __syncthreads();
  const int idx = blockIdx.x * 256 + threadIdx.x;  // kB*kHW*8
  const int g2 = idx & 7;
  const int pos = idx >> 3;
  const int hw = pos & (kHW - 1);
  const int b = pos >> 14;
  const int y = hw >> 7, x = hw & 127;
  float acc[12];
#pragma unroll
  for (int o = 0; o < 12; o++) acc[o] = 0.f;
  for (int k = 0; k < 9; k++) {
    const int ky = k / 3, kx = k - (k / 3) * 3;
    for (int i = 0; i < 12; i++) {
      const int c = g2 * 12 + i;
      const int g = c >> 3;
      const int c2 = 2 * (g * 9 + k);
      const int ddy = c2 / 108, gsy = c2 - ddy * 108;
      const int c2x = c2 + 1;
      const int ddx = c2x / 108, gsx = c2x - ddx * 108;
      const float dy = ref[(((size_t)b * 108 + gsy) * 2 + (1 - ddy)) * kHW + hw];
      const float dx = ref[(((size_t)b * 108 + gsx) * 2 + (1 - ddx)) * kHW + hw];
      const float ml = offm[((size_t)b * 324 + 216 + g * 9 + k) * kHW + hw];
      const float mask = 1.f / (1.f + expf(-ml));
      const float py = (float)(y - 1 + ky) + dy;
      const float px = (float)(x - 1 + kx) + dx;
      const float s = bilin(feat + ((size_t)b * 96 + c) * kHW, py, px) * mask;
      const float* wrow = &ws[(g2 * 12) * 108 + i * 9 + k];
#pragma unroll
      for (int o = 0; o < 12; o++) acc[o] += s * wrow[o * 108];
    }
  }
  for (int o = 0; o < 12; o++) {
    const int oc = g2 * 12 + o;
    q[((size_t)b * 96 + oc) * kHW + hw] = acc[o] + bd[oc];
  }
}

// kv gather: kv[b][c][side*9+s][n] (bf16)
__global__ void k_kv(const float* __restrict__ feat, const float* __restrict__ ref,
                     __bf16* __restrict__ kv, int side) {
  const int i = blockIdx.x * 256 + threadIdx.x;    // kB*12*9*kHW
  const int hw = i & (kHW - 1);
  int t = i >> 14;
  const int s = t % 9; t /= 9;
  const int g = t % 12, b = t / 12;
  const float r0 = ref[(((size_t)b * 108 + g * 9 + s) * 2 + 0) * kHW + hw];
  const float r1 = ref[(((size_t)b * 108 + g * 9 + s) * 2 + 1) * kHW + hw];
  const int y = hw >> 7, x = hw & 127;
  const float gx = (-1.f + 2.f * (float)x / 127.f) + r0 / 127.f * 0.5f;
  const float gy = (-1.f + 2.f * (float)y / 127.f) + r1 / 63.5f;
  const float px = (gx + 1.f) * 0.5f * 127.f;
  const float py = (gy + 1.f) * 0.5f * 127.f;
  for (int ci = 0; ci < 8; ci++) {
    const int c = g * 8 + ci;
    const float v = bilin(feat + ((size_t)b * 96 + c) * kHW, py, px);
    kv[((((size_t)b * 96 + c) * 18) + side * 9 + s) * kHW + hw] = (__bf16)v;
  }
}

__global__ void k_softmax18(float* __restrict__ a) {
  const int i = blockIdx.x * 256 + threadIdx.x;    // kB*12*kHW
  const int hw = i & (kHW - 1);
  const int t = i >> 14;
  const int nh = t % 12, b = t / 12;
  float* base = a + ((size_t)b * 216 + nh * 18) * kHW + hw;
  float mx = -1e30f;
  for (int p = 0; p < 18; p++) mx = fmaxf(mx, base[(size_t)p * kHW]);
  float sum = 0.f;
  float e[18];
  for (int p = 0; p < 18; p++) { e[p] = expf(base[(size_t)p * kHW] - mx); sum += e[p]; }
  const float inv = 1.f / sum;
  for (int p = 0; p < 18; p++) base[(size_t)p * kHW] = e[p] * inv;
}

__global__ void k_combine(const float* __restrict__ a, const __bf16* __restrict__ v,
                          float* __restrict__ o) {
  const int i = blockIdx.x * 256 + threadIdx.x;    // kB*96*kHW
  const int hw = i & (kHW - 1);
  const int t = i >> 14;
  const int c = t % 96, b = t / 96;
  const int hd = c >> 3;
  float s = 0.f;
  for (int p = 0; p < 18; p++)
    s += a[((size_t)b * 216 + hd * 18 + p) * kHW + hw] *
         (float)v[(((size_t)b * 96 + c) * 18 + p) * kHW + hw];
  o[i] = s;
}

// conv_transpose2d(mv, flow_w, stride2, pad implied) + 2*resize2x(ft)
__global__ void k_convtr(const float* __restrict__ mv, const float* __restrict__ fw,
                         const float* __restrict__ fb, const float* __restrict__ fl,
                         float* __restrict__ out) {
  const int i = blockIdx.x * 256 + threadIdx.x;    // kB*256*256
  const int ox = i & 255;
  const int t = i >> 8;
  const int oy = t & 255;
  const int b = t >> 8;
  float a0 = fb[0], a1 = fb[1];
  for (int kh = 0; kh < 4; kh++) {
    const int ty = oy + kh - 2;
    if (ty < 0 || ty > 254 || (ty & 1)) continue;
    const int iy = ty >> 1;
    for (int kw = 0; kw < 4; kw++) {
      const int tx = ox + kw - 2;
      if (tx < 0 || tx > 254 || (tx & 1)) continue;
      const int ix = tx >> 1;
      const float* mp = mv + (size_t)b * 96 * kHW + iy * kW + ix;
      const int wb = (3 - kh) * 4 + (3 - kw);
      for (int c = 0; c < 96; c++) {
        const float m = mp[(size_t)c * kHW];
        a0 += m * fw[c * 32 + wb];
        a1 += m * fw[c * 32 + 16 + wb];
      }
    }
  }
  const float sy = (oy + 0.5f) * 0.5f - 0.5f;
  const float sx = (ox + 0.5f) * 0.5f - 0.5f;
  const float y0f = floorf(sy), x0f = floorf(sx);
  const float wy = sy - y0f, wx = sx - x0f;
  int y0 = (int)y0f, x0 = (int)x0f;
  const int y0c = min(max(y0, 0), 127), y1c = min(max(y0 + 1, 0), 127);
  const int x0c = min(max(x0, 0), 127), x1c = min(max(x0 + 1, 0), 127);
  const float* f0 = fl + (size_t)b * 2 * kHW;
  const float* f1 = f0 + kHW;
  const float r0 = f0[y0c * kW + x0c] * (1.f - wy) * (1.f - wx) + f0[y0c * kW + x1c] * (1.f - wy) * wx +
                   f0[y1c * kW + x0c] * wy * (1.f - wx) + f0[y1c * kW + x1c] * wy * wx;
  const float r1 = f1[y0c * kW + x0c] * (1.f - wy) * (1.f - wx) + f1[y0c * kW + x1c] * (1.f - wy) * wx +
                   f1[y1c * kW + x0c] * wy * (1.f - wx) + f1[y1c * kW + x1c] * wy * wx;
  out[((size_t)b * 2 + 0) * 65536 + oy * 256 + ox] = a0 + 2.f * r0;
  out[((size_t)b * 2 + 1) * 65536 + oy * 256 + ox] = a1 + 2.f * r1;
}

// --------------------------------------------------------------- launcher ---
extern "C" void kernel_launch(void* const* d_in, const int* in_sizes, int n_in,
                              void* d_out, int out_size, void* d_ws, size_t ws_size,
                              hipStream_t stream) {
  (void)in_sizes; (void)n_in; (void)out_size; (void)ws_size;
  // Inputs: feat_t, feat0, feat1, ft0, ft1, then params leaves in sorted-key
  // tree-flatten order (dicts alphabetical, (w,b) tuples in order).
  const float* feat_t = (const float*)d_in[0];
  const float* feat0  = (const float*)d_in[1];
  const float* feat1  = (const float*)d_in[2];
  const float* ft0    = (const float*)d_in[3];
  const float* ft1    = (const float*)d_in[4];
  const float* attn_out_w = (const float*)d_in[5];
  const float* attn_out_b = (const float*)d_in[6];
  const float* attn_w_w   = (const float*)d_in[7];
  const float* attn_w_b   = (const float*)d_in[8];
  const float* blender_w  = (const float*)d_in[9];
  const float* blender_b  = (const float*)d_in[10];
  const float* crf1_w     = (const float*)d_in[11];
  const float* crf1_b     = (const float*)d_in[12];
  const float* crf2_w     = (const float*)d_in[13];
  const float* crf2_b     = (const float*)d_in[14];
  const float* dcn_w      = (const float*)d_in[15];
  const float* dcn_b      = (const float*)d_in[16];
  const float* fc1_w      = (const float*)d_in[17];
  const float* fc1_b      = (const float*)d_in[18];
  const float* fc2_w      = (const float*)d_in[19];
  const float* fc2_b      = (const float*)d_in[20];
  const float* flow_b     = (const float*)d_in[21];
  const float* flow_w     = (const float*)d_in[22];
  const float* offmask_w  = (const float*)d_in[23];
  const float* offmask_b  = (const float*)d_in[24];
  const float* c1_w = (const float*)d_in[25]; const float* c1_b = (const float*)d_in[26];
  const float* c2_w = (const float*)d_in[27]; const float* c2_b = (const float*)d_in[28];
  const float* c3_w = (const float*)d_in[29]; const float* c3_b = (const float*)d_in[30];
  const float* c4_w = (const float*)d_in[31]; const float* c4_b = (const float*)d_in[32];
  const float* c5_w = (const float*)d_in[33]; const float* c5_b = (const float*)d_in[34];
  const float* value_w = (const float*)d_in[35];
  const float* value_b = (const float*)d_in[36];

  // ---- bump allocator over d_ws
  char* wsp = (char*)d_ws;
  auto alloc = [&](size_t bytes) -> void* {
    void* p = (void*)wsp;
    wsp += (bytes + 255) & ~(size_t)255;
    return p;
  };
  const size_t S    = (size_t)kB * 96 * kHW;
  const size_t S192 = (size_t)kB * 192 * kHW;
  const size_t S194 = (size_t)kB * 194 * kHW;
  const size_t S216 = (size_t)kB * 216 * kHW;
  const size_t S288 = (size_t)kB * 288 * kHW;
  const size_t S324 = (size_t)kB * 324 * kHW;
  const size_t SREF = (size_t)kB * 108 * 2 * kHW;
  const size_t SKV  = (size_t)kB * 96 * 18 * kHW;

  float* t_warp = (float*)alloc(S * 4);
  float* t_x    = (float*)alloc(S * 4);
  float* t_mv0  = (float*)alloc(S * 4);
  float* t_mv1  = (float*)alloc(S * 4);
  float* t_off  = (float*)alloc(S324 * 4);
  float* t_ref0 = (float*)alloc(SREF * 4);
  float* t_ref1 = (float*)alloc(SREF * 4);
  float* t_q0   = (float*)alloc(S * 4);
  float* t_q1   = (float*)alloc(S * 4);
  float* t_alog = (float*)alloc(S216 * 4);
  float* t_o    = (float*)alloc(S * 4);
  float* t_att  = (float*)alloc(S * 4);

  __bf16* bf_cat1 = (__bf16*)alloc(S194 * 2);
  __bf16* bf_h1   = (__bf16*)alloc(S192 * 2);
  __bf16* bf_x    = (__bf16*)alloc(S * 2);
  __bf16* bf_r1   = (__bf16*)alloc(S * 2);
  __bf16* bf_r2   = (__bf16*)alloc(S * 2);
  __bf16* bf_r3   = (__bf16*)alloc(S * 2);
  __bf16* bf_r4   = (__bf16*)alloc(S * 2);
  __bf16* bf_mv   = (__bf16*)alloc(S * 2);
  __bf16* bf_qcat = (__bf16*)alloc(S288 * 2);
  __bf16* bf_q    = (__bf16*)alloc(S * 2);
  __bf16* bf_kv   = (__bf16*)alloc(SKV * 2);
  __bf16* bf_v    = (__bf16*)alloc(SKV * 2);
  __bf16* bf_o    = (__bf16*)alloc(S * 2);
  __bf16* bf_att  = (__bf16*)alloc(S * 2);
  __bf16* bf_m    = (__bf16*)alloc(S192 * 2);

  auto pack = [&](const float* w, int M, int K, int Mp, int Kp) -> __bf16* {
    __bf16* dst = (__bf16*)alloc((size_t)Mp * Kp * 2);
    const int tot = Mp * Kp;
    k_pack<<<(tot + 255) / 256, 256, 0, stream>>>(w, dst, M, K, Mp, Kp);
    return dst;
  };
  // conv weights (K = IC*9 rounded up to x32), gemm weights (K already x32)
  __bf16* pw_crf1 = pack(crf1_w, 192, 194 * 9, 192, 1760);
  __bf16* pw_crf2 = pack(crf2_w, 96, 192 * 9, 128, 1728);
  __bf16* pw_c1   = pack(c1_w, 96, 864, 128, 864);
  __bf16* pw_c2   = pack(c2_w, 48, 432, 64, 448);
  __bf16* pw_c3   = pack(c3_w, 96, 864, 128, 864);
  __bf16* pw_c4   = pack(c4_w, 48, 432, 64, 448);
  __bf16* pw_c5   = pack(c5_w, 96, 864, 128, 864);
  __bf16* pw_off  = pack(offmask_w, 324, 864, 384, 864);
  __bf16* pw_bl   = pack(blender_w, 96, 288, 128, 288);
  __bf16* pw_val  = pack(value_w, 96, 96, 128, 96);
  __bf16* pw_aw   = pack(attn_w_w, 216, 96, 256, 96);
  __bf16* pw_ao   = pack(attn_out_w, 96, 96, 128, 96);
  __bf16* pw_fc1  = pack(fc1_w, 192, 96, 192, 96);
  __bf16* pw_fc2  = pack(fc2_w, 96, 192, 128, 192);

  const int NT = kB * kHW;  // 32768 columns for spatial GEMMs
  auto conv3 = [&](const __bf16* Wp, const __bf16* X, long xbs, const float* bias,
                   const float* res, long resbs, float* Of, long obs,
                   __bf16* Oh, long ohbs, int M, int Mp, int IC, int Kp, int act) {
    dim3 g(NT / 128, Mp / 64);
    k_conv3_wmma<<<g, 256, 0, stream>>>(Wp, X, xbs, bias, res, resbs, Of, obs, Oh, ohbs,
                                        M, IC, Kp, act);
  };
  auto gemm = [&](const __bf16* Wp, const __bf16* X, long xbs, long xks, int cpb, int nt,
                  const float* bias, const float* res, long resbs, float* Of, long obs,
                  long oms, __bf16* Oh, long ohbs, int M, int Mp, int K, int act) {
    dim3 g(nt / 128, Mp / 64);
    k_gemm_wmma<<<g, 256, 0, stream>>>(Wp, X, xbs, xks, cpb, bias, res, resbs,
                                       Of, obs, oms, Oh, ohbs, M, K, K, act);
  };

  // ------------------------------ per-branch: movement -> ref -> deform ----
  for (int side = 0; side < 2; side++) {
    const float* feat_s = side ? feat1 : feat0;
    const float* ft_s   = side ? ft1 : ft0;
    float* t_mv_s  = side ? t_mv1 : t_mv0;
    float* t_ref_s = side ? t_ref1 : t_ref0;
    float* t_q_s   = side ? t_q1 : t_q0;

    k_bwarp<<<(int)(S / 256), 256, 0, stream>>>(feat_s, ft_s, t_warp);
    k_cat1<<<(int)(S194 / 256), 256, 0, stream>>>(feat_t, t_warp, ft_s, bf_cat1);
    // crf1: 194ch -> 192ch, prelu (bf16 out only)
    conv3(pw_crf1, bf_cat1, 194 * kHW, crf1_b, nullptr, 0, nullptr, 0,
          bf_h1, 192 * kHW, 192, 192, 194, 1760, 1);
    // crf2: 192 -> 96, prelu (f32 kept for residual)
    conv3(pw_crf2, bf_h1, 192 * kHW, crf2_b, nullptr, 0, t_x, 96 * kHW,
          bf_x, 96 * kHW, 96, 128, 192, 1728, 1);
    // resblock
    conv3(pw_c1, bf_x, 96 * kHW, c1_b, nullptr, 0, nullptr, 0,
          bf_r1, 96 * kHW, 96, 128, 96, 864, 1);
    k_copych48<<<(int)((size_t)kB * 48 * kHW / 256), 256, 0, stream>>>(bf_r1, bf_r2);
    conv3(pw_c2, bf_r1 + 48 * kHW, 96 * kHW, c2_b, nullptr, 0, nullptr, 0,
          bf_r2 + 48 * kHW, 96 * kHW, 48, 64, 48, 448, 1);
    conv3(pw_c3, bf_r2, 96 * kHW, c3_b, nullptr, 0, nullptr, 0,
          bf_r3, 96 * kHW, 96, 128, 96, 864, 1);
    k_copych48<<<(int)((size_t)kB * 48 * kHW / 256), 256, 0, stream>>>(bf_r3, bf_r4);
    conv3(pw_c4, bf_r3 + 48 * kHW, 96 * kHW, c4_b, nullptr, 0, nullptr, 0,
          bf_r4 + 48 * kHW, 96 * kHW, 48, 64, 48, 448, 1);
    // c5: residual + prelu -> mv (f32 + bf16)
    conv3(pw_c5, bf_r4, 96 * kHW, c5_b, t_x, 96 * kHW, t_mv_s, 96 * kHW,
          bf_mv, 96 * kHW, 96, 128, 96, 864, 1);
    // offmask: 96 -> 324, no act
    conv3(pw_off, bf_mv, 96 * kHW, offmask_b, nullptr, 0, t_off, 324 * kHW,
          nullptr, 0, 324, 384, 96, 864, 0);
    // ref coords
    k_ref<<<(int)(SREF / 256), 256, 0, stream>>>(t_off, ft_s, t_ref_s);
    // deformable conv -> q
    k_deform<<<(int)((size_t)kB * kHW * 8 / 256), 256, 0, stream>>>(
        feat_s, t_ref_s, t_off, dcn_w, dcn_b, t_q_s);
  }

  // ------------------------------ attention --------------------------------
  k_qcat<<<(int)(S288 / 256), 256, 0, stream>>>(t_q0, feat_t, t_q1, bf_qcat);
  // blender: 288 -> 96 (bf16 q only)
  gemm(pw_bl, bf_qcat, 288 * kHW, kHW, kHW, NT, blender_b, nullptr, 0,
       nullptr, 0, kHW, bf_q, 96 * kHW, 96, 128, 288, 0);
  // kv gathers (side 0 -> p 0..8, side 1 -> p 9..17)
  k_kv<<<(int)((size_t)kB * 12 * 9 * kHW / 256), 256, 0, stream>>>(feat0, t_ref0, bf_kv, 0);
  k_kv<<<(int)((size_t)kB * 12 * 9 * kHW / 256), 256, 0, stream>>>(feat1, t_ref1, bf_kv, 1);
  // value: v[b][o][p][n] = Wv @ kv  (GEMM over (b,p,n) columns, bf16 out)
  gemm(pw_val, bf_kv, (long)96 * 18 * kHW, (long)18 * kHW, 18 * kHW, kB * 18 * kHW,
       value_b, nullptr, 0, nullptr, 0, (long)18 * kHW, bf_v, (long)96 * 18 * kHW,
       96, 128, 96, 0);
  // attention logits: 96 -> 216 (f32)
  gemm(pw_aw, bf_q, 96 * kHW, kHW, kHW, NT, attn_w_b, nullptr, 0,
       t_alog, 216 * kHW, kHW, nullptr, 0, 216, 256, 96, 0);
  k_softmax18<<<(int)((size_t)kB * 12 * kHW / 256), 256, 0, stream>>>(t_alog);
  k_combine<<<(int)(S / 256), 256, 0, stream>>>(t_alog, bf_v, t_o);
  k_f2bf<<<(int)(S / 256), 256, 0, stream>>>(t_o, bf_o, (int)S);
  // attn_out: 96 -> 96 (f32 att kept as residual, bf16 for MLP)
  gemm(pw_ao, bf_o, 96 * kHW, kHW, kHW, NT, attn_out_b, nullptr, 0,
       t_att, 96 * kHW, kHW, bf_att, 96 * kHW, 96, 128, 96, 0);
  // MLP: fc1 (gelu) then fc2 + residual -> d_out (out tensor)
  gemm(pw_fc1, bf_att, 96 * kHW, kHW, kHW, NT, fc1_b, nullptr, 0,
       nullptr, 0, kHW, bf_m, 192 * kHW, 192, 192, 96, 2);
  float* out_main = (float*)d_out;
  gemm(pw_fc2, bf_m, 192 * kHW, kHW, kHW, NT, fc2_b, t_att, 96 * kHW,
       out_main, 96 * kHW, kHW, nullptr, 0, 96, 128, 192, 0);

  // ------------------------------ flow upsampling --------------------------
  float* up0 = out_main + (size_t)kB * 96 * kHW;
  float* up1 = up0 + (size_t)kB * 2 * 256 * 256;
  k_convtr<<<(int)((size_t)kB * 256 * 256 / 256), 256, 0, stream>>>(t_mv0, flow_w, flow_b, ft0, up0);
  k_convtr<<<(int)((size_t)kB * 256 * 256 / 256), 256, 0, stream>>>(t_mv1, flow_w, flow_b, ft1, up1);
}
